// MBlock_13486197310030
// MI455X (gfx1250) — matrix operand order repaired and checked by reading, not compiled, not run on hardware
//
#include <hip/hip_runtime.h>

// ---------------------------------------------------------------------------
// MI455X / gfx1250 fused transformer block.
// Matrix math: V_WMMA_F32_16X16X32_BF16 (wave32).
// Tile staging:  GLOBAL_LOAD_ASYNC_TO_LDS_B128 (ASYNCcnt) double-buffered.
// ---------------------------------------------------------------------------

typedef __attribute__((ext_vector_type(16))) __bf16 v16bf;
typedef __attribute__((ext_vector_type(8)))  __bf16 bf16x8;
typedef __attribute__((ext_vector_type(8)))  float  v8f;

constexpr int BATCH = 2;
constexpr int T   = 2048;
constexpr int D   = 1024;
constexpr int DC  = 512;
constexpr int DM  = 1536;
constexpr int H   = 16;
constexpr int DE  = 4096;
constexpr int DG  = 1024;
constexpr int CHD = DC / H;    // 32
constexpr int MHD = DM / H;    // 96
constexpr int BT  = BATCH * T; // 4096 rows

// --- CDNA5 async global->LDS copy (16B per lane), tracked by ASYNCcnt ------
__device__ __forceinline__ void async_b128(void* lds_ptr, const void* gptr) {
  // In the LDS aperture the low 32 bits of a generic pointer are the LDS
  // byte address (ISA 10.2).
  const unsigned int lds_off = (unsigned int)(unsigned long long)lds_ptr;
  asm volatile("global_load_async_to_lds_b128 %0, %1, off"
               :: "v"(lds_off), "v"((unsigned long long)gptr)
               : "memory");
}
__device__ __forceinline__ void wait_async0() {
  asm volatile("s_wait_asynccnt 0x0" ::: "memory");
}

// ---------------------------------------------------------------------------
// Transpose + convert: in f32 [K][N]  ->  out bf16 [N][K]
// ---------------------------------------------------------------------------
__global__ __launch_bounds__(256) void cvt_transpose_kernel(
    const float* __restrict__ in, __bf16* __restrict__ out, int K, int N) {
  __shared__ float tile[32][33];
  const int n0 = blockIdx.x * 32;
  const int k0 = blockIdx.y * 32;
  const int tx = threadIdx.x;   // 0..31
  const int ty = threadIdx.y;   // 0..7
#pragma unroll
  for (int j = ty; j < 32; j += 8)
    tile[j][tx] = in[(size_t)(k0 + j) * N + n0 + tx];
  __syncthreads();
#pragma unroll
  for (int j = ty; j < 32; j += 8)
    out[(size_t)(n0 + j) * K + k0 + tx] = (__bf16)tile[tx][j];
}

// ---------------------------------------------------------------------------
// LayerNorm over D=1024, one row per block, writes bf16.
// ---------------------------------------------------------------------------
__global__ __launch_bounds__(256) void ln_kernel(
    const float* __restrict__ x, const float* __restrict__ w,
    const float* __restrict__ b, __bf16* __restrict__ out) {
  const int row = blockIdx.x;
  const int tid = threadIdx.x;
  const float4 xv = ((const float4*)(x + (size_t)row * D))[tid];

  __shared__ float rs[256];
  __shared__ float rq[256];
  rs[tid] = xv.x + xv.y + xv.z + xv.w;
  rq[tid] = xv.x * xv.x + xv.y * xv.y + xv.z * xv.z + xv.w * xv.w;
  __syncthreads();
  for (int off = 128; off > 0; off >>= 1) {
    if (tid < off) { rs[tid] += rs[tid + off]; rq[tid] += rq[tid + off]; }
    __syncthreads();
  }
  const float mu  = rs[0] * (1.0f / D);
  const float var = rq[0] * (1.0f / D) - mu * mu;
  const float inv = rsqrtf(var + 1e-5f);

  const float4 wv = ((const float4*)w)[tid];
  const float4 bv = ((const float4*)b)[tid];
  __bf16* op = out + (size_t)row * D + tid * 4;
  op[0] = (__bf16)((xv.x - mu) * inv * wv.x + bv.x);
  op[1] = (__bf16)((xv.y - mu) * inv * wv.y + bv.y);
  op[2] = (__bf16)((xv.z - mu) * inv * wv.z + bv.z);
  op[3] = (__bf16)((xv.w - mu) * inv * wv.w + bv.w);
}

// ---------------------------------------------------------------------------
// Tiled WMMA GEMM: C[M,N] = A[M,K](bf16) * Bt[N,K](bf16, pre-transposed).
// 256 threads = 8 waves, tile 128x128, K step 32, double-buffered LDS filled
// by async global->LDS loads. Wave grid 4(m) x 2(n): 2x4 WMMA tiles each.
// EPI: 0 = +bias -> bf16
//      1 = silu(+bias) -> bf16
//      2 = z = e*sigmoid(+bias); silu(z) -> bf16        (aux = e, bf16 [M,N])
//      3 = +bias +residual -> f32                       (aux = res, f32 [M,N])
//      4 = +bias -> bf16, stored TRANSPOSED: out[col*M + row] (b128 stores)
// ---------------------------------------------------------------------------
template <int EPI>
__global__ __launch_bounds__(256) void gemm_bf16_kernel(
    const __bf16* __restrict__ A, const __bf16* __restrict__ Bt,
    const float* __restrict__ bias, const void* __restrict__ aux,
    void* __restrict__ outp, int M, int N, int K) {
  __shared__ __bf16 sA[2][128][32];   // A tile  [m][k]
  __shared__ __bf16 sB[2][128][32];   // Bt tile [n][k]

  const int tid  = threadIdx.x;
  const int lane = tid & 31;
  const int wid  = tid >> 5;
  const int wave_m = wid >> 1;      // 0..3
  const int wave_n = wid & 1;       // 0..1
  const int m0 = blockIdx.y * 128;
  const int n0 = blockIdx.x * 128;
  const int nI   = lane & 15;
  const int half = lane >> 4;

  v8f acc[2][4];
#pragma unroll
  for (int i = 0; i < 2; i++)
#pragma unroll
    for (int j = 0; j < 4; j++)
#pragma unroll
      for (int r = 0; r < 8; r++) acc[i][j][r] = 0.0f;

  // staging geometry: 128 rows x 32 k, 2 threads per row, 16 bf16 each
  const int rowS = tid >> 1;
  const int khS  = (tid & 1) * 16;
  const __bf16* aBase = A  + (size_t)(m0 + rowS) * K + khS;
  const __bf16* bBase = Bt + (size_t)(n0 + rowS) * K + khS;

  // prologue: async-stage K-step 0 into buffer 0
  async_b128(&sA[0][rowS][khS],     aBase);
  async_b128(&sA[0][rowS][khS + 8], aBase + 8);
  async_b128(&sB[0][rowS][khS],     bBase);
  async_b128(&sB[0][rowS][khS + 8], bBase + 8);
  wait_async0();
  __syncthreads();

  const int nk = K >> 5;
  for (int ki = 0; ki < nk; ki++) {
    const int cur = ki & 1;
    const int nxt = cur ^ 1;
    // async-prefetch next K-step straight into the other LDS buffer
    if (ki + 1 < nk) {
      const __bf16* ap = aBase + (size_t)(ki + 1) * 32;
      const __bf16* bp = bBase + (size_t)(ki + 1) * 32;
      async_b128(&sA[nxt][rowS][khS],     ap);
      async_b128(&sA[nxt][rowS][khS + 8], ap + 8);
      async_b128(&sB[nxt][rowS][khS],     bp);
      async_b128(&sB[nxt][rowS][khS + 8], bp + 8);
    }

    // A fragments (16x32): lane row = nI; K = half*16 + i (contiguous)
    v16bf af[2];
#pragma unroll
    for (int mt = 0; mt < 2; mt++) {
      const int m   = wave_m * 32 + mt * 16 + nI;
      af[mt] = *(const v16bf*)&sA[cur][m][half * 16];
    }
    // B fragments (32x16): lane col = nI; K = half*16 + i (contiguous)
    v16bf bfr[4];
#pragma unroll
    for (int nt = 0; nt < 4; nt++) {
      const int n = wave_n * 64 + nt * 16 + nI;
      bfr[nt] = *(const v16bf*)&sB[cur][n][half * 16];
    }
#pragma unroll
    for (int mt = 0; mt < 2; mt++)
#pragma unroll
      for (int nt = 0; nt < 4; nt++)
        acc[mt][nt] = __builtin_amdgcn_wmma_f32_16x16x32_bf16(
            false, af[mt], false, bfr[nt], (short)0, acc[mt][nt], false, false);

    wait_async0();      // own async loads into nxt complete
    __syncthreads();    // everyone's are visible
  }

  // epilogue: D layout rows M = half*8 + r, col = nI
#pragma unroll
  for (int mt = 0; mt < 2; mt++) {
#pragma unroll
    for (int nt = 0; nt < 4; nt++) {
      const int col  = n0 + wave_n * 64 + nt * 16 + nI;
      const float bv = bias[col];
      if (EPI == 4) {
        bf16x8 pk;
#pragma unroll
        for (int r = 0; r < 8; r++) pk[r] = (__bf16)(acc[mt][nt][r] + bv);
        const int rowb = m0 + wave_m * 32 + mt * 16 + half * 8;
        *(bf16x8*)((__bf16*)outp + (size_t)col * M + rowb) = pk;
      } else {
#pragma unroll
        for (int r = 0; r < 8; r++) {
          const int row   = m0 + wave_m * 32 + mt * 16 + half * 8 + r;
          const size_t ix = (size_t)row * N + col;
          const float c   = acc[mt][nt][r] + bv;
          if (EPI == 0) {
            ((__bf16*)outp)[ix] = (__bf16)c;
          } else if (EPI == 1) {
            const float sg = 1.0f / (1.0f + __expf(-c));
            ((__bf16*)outp)[ix] = (__bf16)(c * sg);
          } else if (EPI == 2) {
            const float g  = 1.0f / (1.0f + __expf(-c));
            const float z  = (float)((const __bf16*)aux)[ix] * g;
            const float sz = 1.0f / (1.0f + __expf(-z));
            ((__bf16*)outp)[ix] = (__bf16)(z * sz);
          } else {
            ((float*)outp)[ix] = c + ((const float*)aux)[ix];
          }
        }
      }
    }
  }
}

// ---------------------------------------------------------------------------
// Flash attention: grid (T/64, H, B), 128 threads = 4 waves, 16 q rows/wave.
// K and pre-transposed V chunks async-staged into double-buffered LDS.
// p round-trip through wave-private LDS -> wave-local s_wait_dscnt only.
// Causal mask applied only in the <=2 diagonal-straddling chunks per wave.
// ---------------------------------------------------------------------------
__global__ __launch_bounds__(128) void attn_kernel(
    const __bf16* __restrict__ q, const __bf16* __restrict__ k,
    const __bf16* __restrict__ vT, __bf16* __restrict__ y) {
  __shared__ __bf16 sK[2][32][32];     // [key][d]
  __shared__ __bf16 sVt[2][96][32];    // [col][key]
  __shared__ __bf16 sP[4][16][32];     // per-wave p, [qrow][key]

  const int tid  = threadIdx.x;
  const int lane = tid & 31;
  const int wid  = tid >> 5;
  const int qblk = blockIdx.x;
  const int h    = blockIdx.y;
  const int b    = blockIdx.z;
  const int q0   = qblk * 64;
  const int nI   = lane & 15;
  const int half = lane >> 4;
  const int qrow_base = b * T + q0 + wid * 16;

  // Q fragment (A 16x32): lane row = nI, K = half*16 + i (contiguous)
  v16bf qa;
  {
    const __bf16* qp = q + (size_t)(qrow_base + nI) * DC + h * CHD;
    qa = *(const v16bf*)(qp + half * 16);
  }

  v8f acc[6];
#pragma unroll
  for (int ns = 0; ns < 6; ns++)
#pragma unroll
    for (int r = 0; r < 8; r++) acc[ns][r] = 0.0f;
  float mrow[8], lrow[8];
#pragma unroll
  for (int r = 0; r < 8; r++) { mrow[r] = -3e38f; lrow[r] = 0.0f; }

  // staging geometry (16B per lane per matrix slice)
  const int kkS = tid >> 2;            // 0..31 (key row)
  const int d0S = (tid & 3) * 8;       // 0..24 (d / key offset)
  const __bf16* kBase = k + (size_t)(b * T + kkS) * DC + h * CHD + d0S;
  const __bf16* vBase = vT + (size_t)h * MHD * BT + b * T + d0S;  // + col*BT + k0

  const int nchunks = (q0 + 63) / 32 + 1;  // causal: keys 0 .. q0+63 (>= 2)

  // prologue: async-stage chunk 0 into buffer 0
  async_b128(&sK[0][kkS][d0S], kBase);
#pragma unroll
  for (int j = 0; j < 3; j++)
    async_b128(&sVt[0][kkS + 32 * j][d0S], vBase + (size_t)(kkS + 32 * j) * BT);
  wait_async0();
  __syncthreads();

  for (int c = 0; c < nchunks; c++) {
    const int cur = c & 1;
    const int k0  = c * 32;
    // async-prefetch next chunk into the other buffer
    if (c + 1 < nchunks) {
      async_b128(&sK[cur ^ 1][kkS][d0S], kBase + (size_t)(k0 + 32) * DC);
#pragma unroll
      for (int j = 0; j < 3; j++)
        async_b128(&sVt[cur ^ 1][kkS + 32 * j][d0S],
                   vBase + (size_t)(kkS + 32 * j) * BT + k0 + 32);
    }

    // scores: two 16-key tiles, single WMMA each (K = chd = 32)
    v8f zc;
#pragma unroll
    for (int r = 0; r < 8; r++) zc[r] = 0.0f;
    v16bf kb0 = *(const v16bf*)&sK[cur][nI][half * 16];
    v16bf kb1 = *(const v16bf*)&sK[cur][16 + nI][half * 16];
    v8f sc0 = __builtin_amdgcn_wmma_f32_16x16x32_bf16(false, qa, false, kb0,
                                                      (short)0, zc, false, false);
    v8f sc1 = __builtin_amdgcn_wmma_f32_16x16x32_bf16(false, qa, false, kb1,
                                                      (short)0, zc, false, false);

    // online softmax (a row's 16 columns live in one 16-lane half)
    const float scale = 0.17677669529663687f;  // 1/sqrt(32)
    const bool needMask = (k0 + 31) > (q0 + wid * 16);  // wave-uniform
    float alpha[8];
#pragma unroll
    for (int r = 0; r < 8; r++) {
      float s0 = sc0[r] * scale;
      float s1 = sc1[r] * scale;
      if (needMask) {
        const int qr = q0 + wid * 16 + half * 8 + r;
        s0 = (k0 + nI      <= qr) ? s0 : -3e38f;
        s1 = (k0 + 16 + nI <= qr) ? s1 : -3e38f;
      }
      float mx = fmaxf(s0, s1);
      mx = fmaxf(mx, __shfl_xor(mx, 1, 32));
      mx = fmaxf(mx, __shfl_xor(mx, 2, 32));
      mx = fmaxf(mx, __shfl_xor(mx, 4, 32));
      mx = fmaxf(mx, __shfl_xor(mx, 8, 32));
      const float nm = fmaxf(mrow[r], mx);
      const float e0 = __expf(s0 - nm);
      const float e1 = __expf(s1 - nm);
      float ps = e0 + e1;
      ps += __shfl_xor(ps, 1, 32);
      ps += __shfl_xor(ps, 2, 32);
      ps += __shfl_xor(ps, 4, 32);
      ps += __shfl_xor(ps, 8, 32);
      const float al = __expf(mrow[r] - nm);
      lrow[r] = lrow[r] * al + ps;
      mrow[r] = nm;
      alpha[r] = al;
      sP[wid][half * 8 + r][nI]      = (__bf16)e0;
      sP[wid][half * 8 + r][16 + nI] = (__bf16)e1;
    }
#pragma unroll
    for (int ns = 0; ns < 6; ns++)
#pragma unroll
      for (int r = 0; r < 8; r++) acc[ns][r] = acc[ns][r] * alpha[r];

    // sP is wave-private: wave-local DS-counter wait is sufficient.
    asm volatile("s_wait_dscnt 0x0" ::: "memory");

    // p fragment (A 16x32 over keys)
    v16bf pa;
    {
      pa = *(const v16bf*)&sP[wid][nI][half * 16];
    }
    // y += p @ v   (6 column tiles of 16)
#pragma unroll
    for (int ns = 0; ns < 6; ns++) {
      v16bf vb = *(const v16bf*)&sVt[cur][ns * 16 + nI][half * 16];
      acc[ns] = __builtin_amdgcn_wmma_f32_16x16x32_bf16(
          false, pa, false, vb, (short)0, acc[ns], false, false);
    }

    wait_async0();
    __syncthreads();
  }

  // write y (bf16) in [BT, DM] layout
#pragma unroll
  for (int ns = 0; ns < 6; ns++) {
    const int col = h * MHD + ns * 16 + nI;
#pragma unroll
    for (int r = 0; r < 8; r++) {
      const int row = qrow_base + half * 8 + r;
      y[(size_t)row * DM + col] = (__bf16)(acc[ns][r] / lrow[r]);
    }
  }
}

// ---------------------------------------------------------------------------
// Orchestration
// ---------------------------------------------------------------------------
extern "C" void kernel_launch(void* const* d_in, const int* in_sizes, int n_in,
                              void* d_out, int out_size, void* d_ws,
                              size_t ws_size, hipStream_t stream) {
  (void)in_sizes; (void)n_in; (void)out_size; (void)ws_size;
  const float* x    = (const float*)d_in[0];
  // d_in[1] = mask: causal mask implemented analytically, not read.
  const float* ln1w = (const float*)d_in[2];
  const float* ln1b = (const float*)d_in[3];
  const float* wq = (const float*)d_in[4];  const float* bq = (const float*)d_in[5];
  const float* wk = (const float*)d_in[6];  const float* bk = (const float*)d_in[7];
  const float* wv = (const float*)d_in[8];  const float* bv = (const float*)d_in[9];
  const float* wo = (const float*)d_in[10]; const float* bo = (const float*)d_in[11];
  const float* ln2w = (const float*)d_in[12]; const float* ln2b = (const float*)d_in[13];
  const float* we = (const float*)d_in[14]; const float* be = (const float*)d_in[15];
  const float* wg = (const float*)d_in[16]; const float* bg = (const float*)d_in[17];
  const float* wu = (const float*)d_in[18]; const float* bu = (const float*)d_in[19];
  const float* wc = (const float*)d_in[20]; const float* bc = (const float*)d_in[21];
  float* out = (float*)d_out;

  char* p = (char*)d_ws;
  auto carve = [&](size_t elems, size_t esz) -> char* {
    char* r = p;
    p += (elems * esz + 255) & ~(size_t)255;
    return r;
  };
  __bf16* x1   = (__bf16*)carve((size_t)BT * D, 2);
  __bf16* wqb  = (__bf16*)carve((size_t)D * DC, 2);   // [DC][D]
  __bf16* wkb  = (__bf16*)carve((size_t)D * DC, 2);
  __bf16* wvb  = (__bf16*)carve((size_t)D * DM, 2);   // [DM][D]
  __bf16* wob  = (__bf16*)carve((size_t)DM * D, 2);   // [D][DM]
  __bf16* web  = (__bf16*)carve((size_t)D * DE, 2);   // [DE][D]
  __bf16* wgb  = (__bf16*)carve((size_t)D * DG, 2);   // [DG][D]
  __bf16* wub  = (__bf16*)carve((size_t)DG * DE, 2);  // [DE][DG]
  __bf16* wcb  = (__bf16*)carve((size_t)DE * D, 2);   // [D][DE]
  __bf16* qb   = (__bf16*)carve((size_t)BT * DC, 2);
  __bf16* kb   = (__bf16*)carve((size_t)BT * DC, 2);
  __bf16* vTb  = (__bf16*)carve((size_t)BT * DM, 2);  // transposed [DM][BT]
  __bf16* yb   = (__bf16*)carve((size_t)BT * DM, 2);
  float*  xmid = (float*) carve((size_t)BT * D, 4);
  __bf16* x2   = (__bf16*)carve((size_t)BT * D, 2);
  __bf16* eb   = (__bf16*)carve((size_t)BT * DE, 2);
  __bf16* hb   = (__bf16*)carve((size_t)BT * DG, 2);
  __bf16* actb = (__bf16*)carve((size_t)BT * DE, 2);

  const dim3 blk256(256);
  const dim3 blkT(32, 8);

  // LN1 + weight convert/transpose ([K][N] f32 -> [N][K] bf16)
  ln_kernel<<<BT, blk256, 0, stream>>>(x, ln1w, ln1b, x1);
  cvt_transpose_kernel<<<dim3(DC / 32, D / 32),  blkT, 0, stream>>>(wq, wqb, D,  DC);
  cvt_transpose_kernel<<<dim3(DC / 32, D / 32),  blkT, 0, stream>>>(wk, wkb, D,  DC);
  cvt_transpose_kernel<<<dim3(DM / 32, D / 32),  blkT, 0, stream>>>(wv, wvb, D,  DM);
  cvt_transpose_kernel<<<dim3(D / 32,  DM / 32), blkT, 0, stream>>>(wo, wob, DM, D);
  cvt_transpose_kernel<<<dim3(DE / 32, D / 32),  blkT, 0, stream>>>(we, web, D,  DE);
  cvt_transpose_kernel<<<dim3(DG / 32, D / 32),  blkT, 0, stream>>>(wg, wgb, D,  DG);
  cvt_transpose_kernel<<<dim3(DE / 32, DG / 32), blkT, 0, stream>>>(wu, wub, DG, DE);
  cvt_transpose_kernel<<<dim3(D / 32,  DE / 32), blkT, 0, stream>>>(wc, wcb, DE, D);

  // QKV projections (V written transposed for attention staging)
  gemm_bf16_kernel<0><<<dim3(DC / 128, BT / 128), blk256, 0, stream>>>(
      x1, wqb, bq, nullptr, qb, BT, DC, D);
  gemm_bf16_kernel<0><<<dim3(DC / 128, BT / 128), blk256, 0, stream>>>(
      x1, wkb, bk, nullptr, kb, BT, DC, D);
  gemm_bf16_kernel<4><<<dim3(DM / 128, BT / 128), blk256, 0, stream>>>(
      x1, wvb, bv, nullptr, vTb, BT, DM, D);

  // attention
  attn_kernel<<<dim3(T / 64, H, BATCH), dim3(128), 0, stream>>>(qb, kb, vTb, yb);

  // output projection + residual -> xmid (f32)
  gemm_bf16_kernel<3><<<dim3(D / 128, BT / 128), blk256, 0, stream>>>(
      yb, wob, bo, x, xmid, BT, D, DM);

  // LN2
  ln_kernel<<<BT, blk256, 0, stream>>>(xmid, ln2w, ln2b, x2);

  // FFN: e = x2@we + be (bf16)
  gemm_bf16_kernel<0><<<dim3(DE / 128, BT / 128), blk256, 0, stream>>>(
      x2, web, be, nullptr, eb, BT, DE, D);
  // h = silu(x2@wg + bg)
  gemm_bf16_kernel<1><<<dim3(DG / 128, BT / 128), blk256, 0, stream>>>(
      x2, wgb, bg, nullptr, hb, BT, DG, D);
  // act = silu(e * sigmoid(h@wu + bu))
  gemm_bf16_kernel<2><<<dim3(DE / 128, BT / 128), blk256, 0, stream>>>(
      hb, wub, bu, eb, actb, BT, DE, DG);
  // out = xmid + act@wc + bc (f32)
  gemm_bf16_kernel<3><<<dim3(D / 128, BT / 128), blk256, 0, stream>>>(
      actb, wcb, bc, xmid, out, BT, D, DE);
}
